// SparseSpatialAttention2D_16174846837204
// MI455X (gfx1250) — matrix operand order repaired and checked by reading, not compiled, not run on hardware
//
#include <hip/hip_runtime.h>
#include <hip/hip_bf16.h>
#include <math.h>

typedef __attribute__((ext_vector_type(16))) _Float16 v16h;
typedef __attribute__((ext_vector_type(8)))  float    v8f;

#define Bdim 4
#define Hdim 496
#define Wdim 432
#define Npts 400000
#define Cch  128
#define KW   7
#define PADR 3

// ---------------------------------------------------------------------------
// Kernel 1: grid init to -1
// ---------------------------------------------------------------------------
__global__ void k_init_grid(int* __restrict__ grid) {
    int i = blockIdx.x * blockDim.x + threadIdx.x;
    if (i < Bdim * Hdim * Wdim) grid[i] = -1;
}

// ---------------------------------------------------------------------------
// Kernel 2: scatter point ids into grid
// ---------------------------------------------------------------------------
__global__ void k_scatter(const int* __restrict__ idx, int* __restrict__ grid) {
    int n = blockIdx.x * blockDim.x + threadIdx.x;
    if (n >= Npts) return;
    int b = idx[n * 3 + 0];
    int y = idx[n * 3 + 1];
    int x = idx[n * 3 + 2];
    grid[(b * Hdim + y) * Wdim + x] = n;
}

// ---------------------------------------------------------------------------
// Kernel 3: per-point channel mean & max (one wave32 per point, float4 loads)
// ---------------------------------------------------------------------------
__global__ __launch_bounds__(256)
void k_cat(const float* __restrict__ feat, float2* __restrict__ cat) {
    int wid  = (blockIdx.x * blockDim.x + threadIdx.x) >> 5;  // wave id = point id
    int lane = threadIdx.x & 31;
    if (wid >= Npts) return;
    const float4* row = (const float4*)(feat + (size_t)wid * Cch);
    float4 v = row[lane];                       // 32 lanes * 4 = 128 channels
    float s = v.x + v.y + v.z + v.w;
    float m = fmaxf(fmaxf(v.x, v.y), fmaxf(v.z, v.w));
    #pragma unroll
    for (int off = 16; off > 0; off >>= 1) {
        s += __shfl_xor(s, off, 32);
        m  = fmaxf(m, __shfl_xor(m, off, 32));
    }
    if (lane == 0) cat[wid] = make_float2(s * (1.0f / 128.0f), m);
}

// ---------------------------------------------------------------------------
// Kernel 4: fused neighborhood gather + WMMA conv + sigmoid * features
// 8 waves/block, each wave owns a 16-point tile.
// G tile: 16 rows (points) x 128 f16 cols (98 valid: [k*2+c], zero padded).
// conv = G(16x128,f16) x Wvec via 4x v_wmma_f32_16x16x32_f16 (B = weight
// broadcast to all 16 columns -> every column of D equals conv[m]).
// ---------------------------------------------------------------------------
__global__ __launch_bounds__(256)
void k_main(const float* __restrict__ feat, const float* __restrict__ wflat,
            const int* __restrict__ idx, const int* __restrict__ grid,
            const float2* __restrict__ cat, float* __restrict__ out) {
    __shared__ _Float16 wlds[128];
    __shared__ _Float16 G[8][16 * 128];
    __shared__ int spb[8][16], spy[8][16], spx[8][16];

    const int tid  = threadIdx.x;
    const int wave = tid >> 5;
    const int lane = tid & 31;
    const int tile = blockIdx.x * 8 + wave;
    const int base = tile * 16;

    // weight vector -> f16 in LDS (shared by all waves), zero padded to 128
    if (tid < 128) {
        float wv = (tid < KW * KW * 2) ? wflat[tid] : 0.0f;
        wlds[tid] = (_Float16)wv;
    }

    // per-tile point coordinates
    if (lane < 16) {
        int n = base + lane;
        spb[wave][lane] = idx[n * 3 + 0];
        spy[wave][lane] = idx[n * 3 + 1];
        spx[wave][lane] = idx[n * 3 + 2];
    }
    // zero this wave's G tile (16*128 halves = 256 int4)
    int4* Gz = (int4*)&G[wave][0];
    #pragma unroll
    for (int i = 0; i < 8; ++i) Gz[lane + i * 32] = make_int4(0, 0, 0, 0);
    __syncthreads();

    // gather phase: 16 points * 49 offsets = 784 items, strided over 32 lanes
    #pragma unroll 1
    for (int i = 0; i < 25; ++i) {
        int it = lane + i * 32;
        if (it < 16 * 49) {
            int m  = it / 49;
            int k  = it - m * 49;
            int ky = k / 7;
            int dy = ky - PADR;
            int dx = (k - ky * 7) - PADR;
            int y  = spy[wave][m] + dy;
            int x  = spx[wave][m] + dx;
            if ((unsigned)y < (unsigned)Hdim && (unsigned)x < (unsigned)Wdim) {
                int nbr = grid[(spb[wave][m] * Hdim + y) * Wdim + x];
                if (nbr >= 0) {
                    float2 cv = cat[nbr];               // L2-resident gather
                    G[wave][m * 128 + 2 * k + 0] = (_Float16)cv.x;
                    G[wave][m * 128 + 2 * k + 1] = (_Float16)cv.y;
                }
            }
        }
    }
    __syncthreads();

    // WMMA phase: D = G * Wbroadcast, f32 accumulate over 4 K-chunks of 32
    const int row = lane & 15;
    const int hi  = lane >> 4;
    const _Float16* Gw = &G[wave][0];
    v8f acc = {};
    #pragma unroll
    for (int kc = 0; kc < 4; ++kc) {
        const int K0 = kc * 32;
        v16h a, b;
        // A 16x32 f16 layout: lanes 0-15 row M=lane, K0+{0..7} then K0+16+{0..7};
        //                     lanes 16-31 same row, K offset +8.
        #pragma unroll
        for (int h = 0; h < 8; ++h) a[h]     = Gw[row * 128 + K0 + hi * 8 + h];
        #pragma unroll
        for (int h = 0; h < 8; ++h) a[8 + h] = Gw[row * 128 + K0 + 16 + hi * 8 + h];
        // B 32x16 f16 layout: lane = column; lanes 0-15 hold K0+{0..7} then K0+16+{0..7},
        //                     lanes 16-31 same, K offset +8. Same weights in
        //                     every column -> all D columns equal conv[m].
        #pragma unroll
        for (int h = 0; h < 16; ++h) b[h] = wlds[K0 + hi * 8 + (h & 7) + (h >> 3) * 16];
        acc = __builtin_amdgcn_wmma_f32_16x16x32_f16(
            /*neg_a=*/false, a, /*neg_b=*/false, b,
            /*c_mod=*/(short)0, acc, /*reuse_a=*/false, /*reuse_b=*/false);
    }

    // writeback: conv[p] lives in acc[p&7] of lane 0 (p<8) / lane 16 (p>=8)
    const float4* f4 = (const float4*)feat;
    float4*       o4 = (float4*)out;
    #pragma unroll
    for (int p = 0; p < 16; ++p) {
        float convp = __shfl(acc[p & 7], (p & 8) ? 16 : 0, 32);
        float sg = 1.0f / (1.0f + __expf(-convp));
        size_t n = (size_t)(base + p);
        float4 f = f4[n * 32 + lane];            // 32 lanes * float4 = 128 ch
        o4[n * 32 + lane] = make_float4(f.x * sg, f.y * sg, f.z * sg, f.w * sg);
    }
}

// ---------------------------------------------------------------------------
extern "C" void kernel_launch(void* const* d_in, const int* in_sizes, int n_in,
                              void* d_out, int out_size, void* d_ws, size_t ws_size,
                              hipStream_t stream) {
    const float* feat  = (const float*)d_in[0];   // (N, 128) f32
    const float* wgt   = (const float*)d_in[1];   // (7, 7, 2) f32
    const int*   idx   = (const int*)d_in[2];     // (N, 3)   i32
    float*       out   = (float*)d_out;           // (N, 128) f32

    // workspace layout: grid ints, then cat float2 (offset is 16B aligned)
    int*    grid = (int*)d_ws;
    float2* cat  = (float2*)((char*)d_ws + (size_t)Bdim * Hdim * Wdim * sizeof(int));

    const int gridElems = Bdim * Hdim * Wdim;                 // 857088
    k_init_grid<<<(gridElems + 255) / 256, 256, 0, stream>>>(grid);
    k_scatter <<<(Npts + 255) / 256, 256, 0, stream>>>(idx, grid);
    k_cat     <<<(Npts * 32 + 255) / 256, 256, 0, stream>>>(feat, cat);   // 50000 blocks
    k_main    <<<Npts / (16 * 8), 256, 0, stream>>>(feat, wgt, idx, grid, cat, out); // 3125 blocks
}
